// LocalDenseCrossReadout_63488206569831
// MI455X (gfx1250) — matrix operand order repaired and checked by reading, not compiled, not run on hardware
//
#include <hip/hip_runtime.h>

// ---------------------------------------------------------------------------
// LocalDenseCrossReadout on MI455X (gfx1250, wave32, WMMA bf16 16x16x32)
// Pipeline:
//   1. transpose weights fp32 -> bf16 (W^T, K-contiguous rows = WMMA B layout)
//   2. FiLM: h = silu(ctx0@Wc0 + ctx1@Wc1 + b), [gamma|beta] = h@Wf + bf
//   3. LayerNorm(+FiLM for q) -> bf16 activations
//   4. bf16 WMMA GEMMs with ASYNC-to-LDS double-buffered A staging:
//      q_p, k_p (row-major bf16), v_p (transposed bf16)
//   5. rank-32 gate GEMMs
//   6. fused local attention (scores + gate bias + softmax + P@V) per
//      (batch, q-step); mask computed analytically (9-step window)
//   7. output projection -> fp32 d_out
// ---------------------------------------------------------------------------

typedef __attribute__((ext_vector_type(16))) __bf16 bf16x16;
typedef __attribute__((ext_vector_type(8)))  __bf16 bf16x8;
typedef __attribute__((ext_vector_type(4)))  __bf16 bf16x4;
typedef __attribute__((ext_vector_type(8)))  float  f32x8;

#define DIM   512
#define NB    8
#define QTOK  1024
#define KTOK  4096
#define RANK  32
#define WIN   4
#define KSTEP 256

__device__ __forceinline__ f32x8 wmma_bf16(bf16x16 a, bf16x16 b, f32x8 c) {
  return __builtin_amdgcn_wmma_f32_16x16x32_bf16(
      /*neg_a=*/false, a, /*neg_b=*/false, b, /*c_mod=*/(short)0, c,
      /*reuse_a=*/false, /*reuse_b=*/false);
}

// Build a 16-element bf16 fragment from two contiguous 8xbf16 (16B) loads.
__device__ __forceinline__ bf16x16 load_ab(const __bf16* p0, const __bf16* p1) {
  bf16x8 lo = *(const bf16x8*)p0;
  bf16x8 hi = *(const bf16x8*)p1;
  bf16x16 r;
#pragma unroll
  for (int i = 0; i < 8; ++i) { r[i] = lo[i]; r[i + 8] = hi[i]; }
  return r;
}

// ---------------------------------------------------------------------------
// Weight transpose + bf16 cast:  WT[n*K + k] = W[k*N + n]
// ---------------------------------------------------------------------------
__global__ __launch_bounds__(256) void transpose_bf16_kernel(
    const float* __restrict__ W, __bf16* __restrict__ WT, int K, int N) {
  int idx = blockIdx.x * 256 + threadIdx.x;
  if (idx >= K * N) return;
  int n = idx / K, k = idx - n * K;
  WT[idx] = (__bf16)W[(size_t)k * N + n];
}

// ---------------------------------------------------------------------------
// FiLM conditioning (tiny: 8x512)
// ---------------------------------------------------------------------------
__global__ __launch_bounds__(256) void film_h_kernel(
    const float* __restrict__ ctx0, const float* __restrict__ ctx1,
    const float* __restrict__ Wc0, const float* __restrict__ bc0,
    const float* __restrict__ Wc1, const float* __restrict__ bc1,
    float* __restrict__ h) {
  int idx = blockIdx.x * 256 + threadIdx.x;
  if (idx >= NB * DIM) return;
  int b = idx >> 9, d = idx & (DIM - 1);
  const float* x0 = ctx0 + b * DIM;
  const float* x1 = ctx1 + b * DIM;
  float acc = bc0[d] + bc1[d];
  for (int e = 0; e < DIM; ++e)
    acc += x0[e] * Wc0[(size_t)e * DIM + d] + x1[e] * Wc1[(size_t)e * DIM + d];
  h[idx] = acc / (1.0f + __expf(-acc));  // silu
}

__global__ __launch_bounds__(256) void film_gb_kernel(
    const float* __restrict__ h, const float* __restrict__ Wf,
    const float* __restrict__ bfv, float* __restrict__ gamma,
    float* __restrict__ beta) {
  int idx = blockIdx.x * 256 + threadIdx.x;
  if (idx >= NB * 2 * DIM) return;
  int b = idx >> 10, j = idx & (2 * DIM - 1);
  const float* hr = h + b * DIM;
  float acc = bfv[j];
  for (int e = 0; e < DIM; ++e) acc += hr[e] * Wf[(size_t)e * 2 * DIM + j];
  if (j < DIM) gamma[b * DIM + j] = acc;
  else         beta[b * DIM + j - DIM] = acc;
}

// ---------------------------------------------------------------------------
// LayerNorm (+ optional FiLM modulation) -> bf16. One wave per 512-wide row.
// ---------------------------------------------------------------------------
__global__ __launch_bounds__(256) void ln_film_kernel(
    const float* __restrict__ X, const float* __restrict__ g,
    const float* __restrict__ bb, const float* __restrict__ gamma,
    const float* __restrict__ beta, __bf16* __restrict__ Y, int rowsPerBatch) {
  int wave = threadIdx.x >> 5, lane = threadIdx.x & 31;
  int row = blockIdx.x * 8 + wave;
  const float4* x4 = (const float4*)(X + (size_t)row * DIM);
  float4 v[4];
  float s = 0.f, s2 = 0.f;
#pragma unroll
  for (int i = 0; i < 4; ++i) {
    v[i] = x4[lane + 32 * i];
    s  += v[i].x + v[i].y + v[i].z + v[i].w;
    s2 += v[i].x * v[i].x + v[i].y * v[i].y + v[i].z * v[i].z + v[i].w * v[i].w;
  }
#pragma unroll
  for (int off = 16; off; off >>= 1) {
    s  += __shfl_xor(s, off, 32);
    s2 += __shfl_xor(s2, off, 32);
  }
  float mu  = s * (1.0f / DIM);
  float inv = rsqrtf(s2 * (1.0f / DIM) - mu * mu + 1e-5f);
  int b = row / rowsPerBatch;
#pragma unroll
  for (int i = 0; i < 4; ++i) {
    int d0 = (lane + 32 * i) * 4;
    float xv[4] = {v[i].x, v[i].y, v[i].z, v[i].w};
    bf16x4 o;
#pragma unroll
    for (int c = 0; c < 4; ++c) {
      int d = d0 + c;
      float y = (xv[c] - mu) * inv * g[d] + bb[d];
      if (gamma) y = y * (1.0f + gamma[b * DIM + d]) + beta[b * DIM + d];
      o[c] = (__bf16)y;
    }
    *(bf16x4*)(Y + (size_t)row * DIM + d0) = o;
  }
}

// ---------------------------------------------------------------------------
// GEMM  C[M,512] = A[M,512](bf16) @ W(=WT rows, bf16) + bias
// Block 256 thr = 8 waves (4x2); block tile 128x64; wave tile 32x32 (2x2 WMMA)
// A tile (128x32 bf16 = 8KB) staged to LDS with GLOBAL_LOAD_ASYNC_TO_LDS_B128,
// double-buffered on ASYNCcnt (async loads complete in order -> wait<=2 keeps
// the prefetch in flight while retiring the current buffer). B (weights) is
// read direct from global: small, L2-resident, contiguous per lane.
// Outputs: fp32 row-major, bf16 row-major, bf16 transposed [b][n][t] (any/all)
// ---------------------------------------------------------------------------
#define AROWS 128
#define ASTR  40  // bf16 elems per LDS row: 80B row stride (16B-aligned, bank-spread)

__global__ __launch_bounds__(256) void gemm512_kernel(
    const __bf16* __restrict__ A, const __bf16* __restrict__ WT,
    const float* __restrict__ bias, float* __restrict__ Cf,
    __bf16* __restrict__ Cb, __bf16* __restrict__ CbT, int TB) {
  const int K = DIM;
  __shared__ __align__(16) __bf16 Alds[2 * AROWS * ASTR];
  int tid = threadIdx.x;
  int lane = tid & 31, wave = tid >> 5;
  int lr = lane & 15, hb = lane >> 4;
  int mblk = blockIdx.x * 128;
  int mw = (wave & 3) * 32;
  int n0 = blockIdx.y * 64 + (wave >> 2) * 32;

  // Async staging: 128 rows x 64B; 512 x 16B chunks; 2 chunks per thread.
  int r0 = tid >> 2, r1 = r0 + 64;      // chunk rows
  int cbyte = (tid & 3) * 16;           // 16B chunk within the 64B row slice
  unsigned lbase = (unsigned)(size_t)(&Alds[0]);
  const char* Ab = (const char*)A;

  auto stage = [&](int buf, int kk) {
    const char* g0 = Ab + ((size_t)(mblk + r0) * K + kk) * 2 + cbyte;
    const char* g1 = Ab + ((size_t)(mblk + r1) * K + kk) * 2 + cbyte;
    unsigned l0 = lbase + (unsigned)(buf * AROWS * ASTR * 2 + r0 * (ASTR * 2) + cbyte);
    unsigned l1 = l0 + 64 * (ASTR * 2);
    asm volatile("global_load_async_to_lds_b128 %0, %1, off"
                 :: "v"(l0), "v"(g0) : "memory");
    asm volatile("global_load_async_to_lds_b128 %0, %1, off"
                 :: "v"(l1), "v"(g1) : "memory");
  };

  f32x8 c[2][2] = {};
  const __bf16* b0r = WT + (size_t)(n0 + lr) * K;
  const __bf16* b1r = WT + (size_t)(n0 + 16 + lr) * K;

  stage(0, 0);
  for (int kk = 0; kk < K; kk += 32) {
    int buf = (kk >> 5) & 1;
    if (kk + 32 < K) {
      stage(buf ^ 1, kk + 32);
      asm volatile("s_wait_asynccnt 0x2" ::: "memory");
    } else {
      asm volatile("s_wait_asynccnt 0x0" ::: "memory");
    }
    __syncthreads();  // publish current buffer across all waves
    const __bf16* al = Alds + buf * AROWS * ASTR;
    const __bf16* a0p = al + (mw + lr) * ASTR + 8 * hb;
    const __bf16* a1p = al + (mw + 16 + lr) * ASTR + 8 * hb;
    bf16x16 a0 = load_ab(a0p, a0p + 16);
    bf16x16 a1 = load_ab(a1p, a1p + 16);
    bf16x16 b0 = load_ab(b0r + kk + 8 * hb, b0r + kk + 16 + 8 * hb);
    bf16x16 b1 = load_ab(b1r + kk + 8 * hb, b1r + kk + 16 + 8 * hb);
    c[0][0] = wmma_bf16(a0, b0, c[0][0]);
    c[0][1] = wmma_bf16(a0, b1, c[0][1]);
    c[1][0] = wmma_bf16(a1, b0, c[1][0]);
    c[1][1] = wmma_bf16(a1, b1, c[1][1]);
    __syncthreads();  // all reads done before buffer is overwritten
  }

#pragma unroll
  for (int mi = 0; mi < 2; ++mi)
#pragma unroll
    for (int ni = 0; ni < 2; ++ni) {
      int col = n0 + ni * 16 + lr;
      float bv = bias ? bias[col] : 0.0f;
      int mr = mblk + mw + mi * 16 + hb * 8;
      f32x8 acc = c[mi][ni];
      if (Cf) {
#pragma unroll
        for (int r = 0; r < 8; ++r)
          Cf[(size_t)(mr + r) * DIM + col] = acc[r] + bv;
      }
      if (Cb) {
#pragma unroll
        for (int r = 0; r < 8; ++r)
          Cb[(size_t)(mr + r) * DIM + col] = (__bf16)(acc[r] + bv);
      }
      if (CbT) {  // [batch][col][token], 8 consecutive tokens per lane
        int bidx = mr / TB, t = mr - bidx * TB;
        bf16x8 o;
#pragma unroll
        for (int r = 0; r < 8; ++r) o[r] = (__bf16)(acc[r] + bv);
        *(bf16x8*)(CbT + ((size_t)bidx * DIM + col) * TB + t) = o;
      }
    }
}

// ---------------------------------------------------------------------------
// Gate GEMM: Out[M,32](bf16) = A[M,512](bf16) @ Wg (WgT rows, bf16)
// 8 waves, each 16 rows x 32 cols (2 WMMA tiles), grid = M/128
// ---------------------------------------------------------------------------
__global__ __launch_bounds__(256) void gemm_gate_kernel(
    const __bf16* __restrict__ A, const __bf16* __restrict__ WT,
    __bf16* __restrict__ Out) {
  const int K = DIM;
  int lane = threadIdx.x & 31, wave = threadIdx.x >> 5;
  int lr = lane & 15, hb = lane >> 4;
  int m0 = blockIdx.x * 128 + wave * 16;
  f32x8 c0 = {}, c1 = {};
  const __bf16* ar  = A + (size_t)(m0 + lr) * K;
  const __bf16* b0r = WT + (size_t)lr * K;
  const __bf16* b1r = WT + (size_t)(16 + lr) * K;
  for (int kk = 0; kk < K; kk += 32) {
    bf16x16 a  = load_ab(ar + kk + 16 * hb, ar + kk + 16 * hb + 8);
    bf16x16 b0 = load_ab(b0r + kk + 16 * hb, b0r + kk + 16 * hb + 8);
    bf16x16 b1 = load_ab(b1r + kk + 16 * hb, b1r + kk + 16 * hb + 8);
    c0 = wmma_bf16(a, b0, c0);
    c1 = wmma_bf16(a, b1, c1);
  }
#pragma unroll
  for (int r = 0; r < 8; ++r) {
    int mr = m0 + hb * 8 + r;
    Out[(size_t)mr * RANK + lr]      = (__bf16)c0[r];
    Out[(size_t)mr * RANK + 16 + lr] = (__bf16)c1[r];
  }
}

// ---------------------------------------------------------------------------
// Fused local attention. Grid (64 q-steps, 8 batches), 256 threads = 8 waves.
// Window: <=9 consecutive kv-steps of 16 tokens; scores 16x(<=144), fp32 LDS,
// gate bias, softmax per row, bf16 P in LDS, P@V over transposed V.
// ---------------------------------------------------------------------------
#define SSTR 168  // padded row stride (>=160 cols, 16B-aligned in bf16)

__global__ __launch_bounds__(256) void attn_kernel(
    const __bf16* __restrict__ qp, const __bf16* __restrict__ kp,
    const __bf16* __restrict__ vpT, const __bf16* __restrict__ gq,
    const __bf16* __restrict__ gk, __bf16* __restrict__ ctx) {
  __shared__ __align__(16) float  Sf[16 * SSTR];
  __shared__ __align__(16) __bf16 Pb[16 * SSTR];
  int qi = blockIdx.x, b = blockIdx.y;
  int tid = threadIdx.x, wave = tid >> 5, lane = tid & 31;
  int lr = lane & 15, hb = lane >> 4;

  int cen = __float2int_rn(qi * (255.0f / 63.0f));  // centers, no .5 ties
  int lo = max(0, cen - WIN);
  int hiS = min(KSTEP - 1, cen + WIN);
  int nsteps = hiS - lo + 1;  // 5..9
  int tok0 = lo * 16;

  for (int i = tid; i < 16 * SSTR; i += 256) Sf[i] = -__builtin_inff();
  __syncthreads();

  const float scale = 0.04419417382415922f;  // 512^-0.5
  const float rinv  = 0.17677669529663687f;  // 32^-0.5
  size_t qbase = (size_t)b * QTOK + (size_t)qi * 16;
  const __bf16* arow  = qp + (qbase + lr) * DIM;
  const __bf16* garow = gq + (qbase + lr) * RANK;

  // Phase 1: score + gate tiles (one 16x16 tile per window kv-step)
  for (int j = wave; j < nsteps; j += 8) {
    int t0 = tok0 + j * 16;
    const __bf16* brow = kp + ((size_t)b * KTOK + t0 + lr) * DIM;
    f32x8 acc = {};
    for (int kk = 0; kk < DIM; kk += 32) {
      bf16x16 a  = load_ab(arow + kk + 16 * hb, arow + kk + 16 * hb + 8);
      bf16x16 bm = load_ab(brow + kk + 16 * hb, brow + kk + 16 * hb + 8);
      acc = wmma_bf16(a, bm, acc);
    }
    const __bf16* gbrow = gk + ((size_t)b * KTOK + t0 + lr) * RANK;
    bf16x16 ga = load_ab(garow + 16 * hb, garow + 16 * hb + 8);
    bf16x16 gb = load_ab(gbrow + 16 * hb, gbrow + 16 * hb + 8);
    f32x8 gacc = {};
    gacc = wmma_bf16(ga, gb, gacc);
#pragma unroll
    for (int r = 0; r < 8; ++r) {
      float gl = gacc[r] * rinv;
      float sg = 1.0f / (1.0f + __expf(-gl));
      Sf[(hb * 8 + r) * SSTR + j * 16 + lr] = acc[r] * scale + __logf(sg + 1e-6f);
    }
  }
  __syncthreads();

  // Phase 2: softmax over 160 padded cols (invalid = -inf -> 0). Wave per row.
  for (int r = wave; r < 16; r += 8) {
    float vv[5];
    float m = -__builtin_inff();
#pragma unroll
    for (int i = 0; i < 5; ++i) {
      vv[i] = Sf[r * SSTR + lane + 32 * i];
      m = fmaxf(m, vv[i]);
    }
#pragma unroll
    for (int off = 16; off; off >>= 1) m = fmaxf(m, __shfl_xor(m, off, 32));
    float ssum = 0.f;
#pragma unroll
    for (int i = 0; i < 5; ++i) { vv[i] = __expf(vv[i] - m); ssum += vv[i]; }
#pragma unroll
    for (int off = 16; off; off >>= 1) ssum += __shfl_xor(ssum, off, 32);
    float inv = 1.0f / ssum;
#pragma unroll
    for (int i = 0; i < 5; ++i)
      Pb[r * SSTR + lane + 32 * i] = (__bf16)(vv[i] * inv);
  }
  __syncthreads();

  // Phase 3: ctx[16,512] = P[16,160] @ V; V columns contiguous in vpT.
  f32x8 acc[4] = {};
  int nb0 = wave * 64;
  for (int kc = 0; kc < 5; ++kc) {
    const __bf16* ap = &Pb[lr * SSTR + kc * 32 + 16 * hb];
    bf16x16 a = load_ab(ap, ap + 8);
    int tB = tok0 + kc * 32 + 16 * hb;
#pragma unroll
    for (int ni = 0; ni < 4; ++ni) {
      int dim = nb0 + ni * 16 + lr;
      const __bf16* vr = vpT + ((size_t)b * DIM + dim) * KTOK;
      bf16x16 bm;
      if (tB + 15 < KTOK) {
        bm = load_ab(vr + tB, vr + tB + 8);
      } else {  // clamp (P==0 in padded region, any finite value is safe)
#pragma unroll
        for (int i = 0; i < 16; ++i) bm[i] = vr[min(tB + i, KTOK - 1)];
      }
      acc[ni] = wmma_bf16(a, bm, acc[ni]);
    }
  }
#pragma unroll
  for (int ni = 0; ni < 4; ++ni) {
    int dim = nb0 + ni * 16 + lr;
#pragma unroll
    for (int r = 0; r < 8; ++r)
      ctx[(qbase + hb * 8 + r) * DIM + dim] = (__bf16)acc[ni][r];
  }
}

// ---------------------------------------------------------------------------
extern "C" void kernel_launch(void* const* d_in, const int* in_sizes, int n_in,
                              void* d_out, int out_size, void* d_ws,
                              size_t ws_size, hipStream_t stream) {
  (void)in_sizes; (void)n_in; (void)out_size; (void)ws_size;
  const float* query  = (const float*)d_in[0];
  const float* source = (const float*)d_in[1];
  const float* ctx0   = (const float*)d_in[2];
  const float* ctx1   = (const float*)d_in[3];
  /* d_in[4] = mask: recomputed analytically (local window), not read */
  const float* qn_g  = (const float*)d_in[5];
  const float* qn_b  = (const float*)d_in[6];
  const float* kvn_g = (const float*)d_in[7];
  const float* kvn_b = (const float*)d_in[8];
  const float* Wq = (const float*)d_in[9];  const float* bq = (const float*)d_in[10];
  const float* Wk = (const float*)d_in[11]; const float* bk = (const float*)d_in[12];
  const float* Wv = (const float*)d_in[13]; const float* bv = (const float*)d_in[14];
  const float* Wo = (const float*)d_in[15]; const float* bo = (const float*)d_in[16];
  const float* Wgq = (const float*)d_in[17];
  const float* Wgk = (const float*)d_in[18];
  const float* Wc0 = (const float*)d_in[19]; const float* bc0 = (const float*)d_in[20];
  const float* Wc1 = (const float*)d_in[21]; const float* bc1 = (const float*)d_in[22];
  const float* Wf  = (const float*)d_in[23]; const float* bfv = (const float*)d_in[24];

  char* wsb = (char*)d_ws;
  size_t off = 0;
  auto alloc = [&](size_t bytes) -> void* {
    void* p = wsb + off;
    off += (bytes + 255) & ~(size_t)255;
    return p;
  };
  float*  h_buf = (float*)alloc((size_t)NB * DIM * 4);
  float*  gamma = (float*)alloc((size_t)NB * DIM * 4);
  float*  beta  = (float*)alloc((size_t)NB * DIM * 4);
  __bf16* WqT   = (__bf16*)alloc((size_t)DIM * DIM * 2);
  __bf16* WkT   = (__bf16*)alloc((size_t)DIM * DIM * 2);
  __bf16* WvT   = (__bf16*)alloc((size_t)DIM * DIM * 2);
  __bf16* WoT   = (__bf16*)alloc((size_t)DIM * DIM * 2);
  __bf16* WgqT  = (__bf16*)alloc((size_t)DIM * RANK * 2);
  __bf16* WgkT  = (__bf16*)alloc((size_t)DIM * RANK * 2);
  __bf16* qmod  = (__bf16*)alloc((size_t)NB * QTOK * DIM * 2);
  __bf16* sbase = (__bf16*)alloc((size_t)NB * KTOK * DIM * 2);
  __bf16* qpb   = (__bf16*)alloc((size_t)NB * QTOK * DIM * 2);
  __bf16* kpb   = (__bf16*)alloc((size_t)NB * KTOK * DIM * 2);
  __bf16* vpT   = (__bf16*)alloc((size_t)NB * DIM * KTOK * 2);
  __bf16* gqb   = (__bf16*)alloc((size_t)NB * QTOK * RANK * 2);
  __bf16* gkb   = (__bf16*)alloc((size_t)NB * KTOK * RANK * 2);
  __bf16* ctxb  = (__bf16*)alloc((size_t)NB * QTOK * DIM * 2);

  // 1. weights -> bf16 transposed
  transpose_bf16_kernel<<<1024, 256, 0, stream>>>(Wq, WqT, DIM, DIM);
  transpose_bf16_kernel<<<1024, 256, 0, stream>>>(Wk, WkT, DIM, DIM);
  transpose_bf16_kernel<<<1024, 256, 0, stream>>>(Wv, WvT, DIM, DIM);
  transpose_bf16_kernel<<<1024, 256, 0, stream>>>(Wo, WoT, DIM, DIM);
  transpose_bf16_kernel<<<64, 256, 0, stream>>>(Wgq, WgqT, DIM, RANK);
  transpose_bf16_kernel<<<64, 256, 0, stream>>>(Wgk, WgkT, DIM, RANK);

  // 2. FiLM conditioning
  film_h_kernel<<<16, 256, 0, stream>>>(ctx0, ctx1, Wc0, bc0, Wc1, bc1, h_buf);
  film_gb_kernel<<<32, 256, 0, stream>>>(h_buf, Wf, bfv, gamma, beta);

  // 3. LayerNorm (+FiLM for q) -> bf16
  ln_film_kernel<<<1024, 256, 0, stream>>>(query, qn_g, qn_b, gamma, beta,
                                           qmod, QTOK);
  ln_film_kernel<<<4096, 256, 0, stream>>>(source, kvn_g, kvn_b, nullptr,
                                           nullptr, sbase, KTOK);

  // 4. projections
  gemm512_kernel<<<dim3(64, 8), 256, 0, stream>>>(qmod, WqT, bq, nullptr, qpb,
                                                  nullptr, 1);
  gemm512_kernel<<<dim3(256, 8), 256, 0, stream>>>(sbase, WkT, bk, nullptr,
                                                   kpb, nullptr, 1);
  gemm512_kernel<<<dim3(256, 8), 256, 0, stream>>>(sbase, WvT, bv, nullptr,
                                                   nullptr, vpT, KTOK);

  // 5. gates
  gemm_gate_kernel<<<64, 256, 0, stream>>>(qpb, WgqT, gqb);
  gemm_gate_kernel<<<256, 256, 0, stream>>>(kpb, WgkT, gkb);

  // 6. fused local attention
  attn_kernel<<<dim3(64, 8), 256, 0, stream>>>(qpb, kpb, vpT, gqb, gkb, ctxb);

  // 7. output projection -> fp32 d_out
  gemm512_kernel<<<dim3(64, 8), 256, 0, stream>>>(ctxb, WoT, bo, (float*)d_out,
                                                  nullptr, nullptr, 1);
}
